// BasicBlock_1709396984498
// MI455X (gfx1250) — matrix operand order repaired and checked by reading, not compiled, not run on hardware
//
#include <hip/hip_runtime.h>

typedef _Float16 half_t;
typedef __attribute__((ext_vector_type(16))) _Float16 v16h;
typedef __attribute__((ext_vector_type(8)))  _Float16 v8h;
typedef __attribute__((ext_vector_type(8)))  float    v8f;

#define GAMMA_1 0.3f
#define GAMMA_2 0.7f

constexpr int  Bn = 8, Hh = 256, Ww = 256;
constexpr int  HW = Hh * Ww;                 // 65536
constexpr long N3 = (long)Bn * 3 * HW;       // 1572864
constexpr long N1 = (long)Bn * HW;           // 524288

// ------------------------------------------------------------------
// per-(b,c) spatial mean of a [B,3,H,W] fp32 tensor -> out[24]
__global__ void means_kernel(const float* __restrict__ x, float* __restrict__ out) {
  __shared__ float sm[256];
  const float* p = x + (long)blockIdx.x * HW;
  float s = 0.f;
  for (int i = threadIdx.x; i < HW; i += 256) s += p[i];
  sm[threadIdx.x] = s; __syncthreads();
  for (int o = 128; o > 0; o >>= 1) {
    if (threadIdx.x < o) sm[threadIdx.x] += sm[threadIdx.x + o];
    __syncthreads();
  }
  if (threadIdx.x == 0) out[blockIdx.x] = sm[0] / (float)HW;
}

// ------------------------------------------------------------------
__device__ __forceinline__ float softthr(float x, float thr) {
  float s = fabsf(x) - thr;
  s = s > 0.f ? s : 0.f;
  return copysignf(s, x);
}

// channel sort/scatter of J + u,v,w1,w2 dual updates
__global__ void scatter_kernel(const float* __restrict__ J, const float* __restrict__ u,
                               const float* __restrict__ v, const float* __restrict__ w1,
                               const float* __restrict__ w2, const float* __restrict__ lam1p,
                               const float* __restrict__ lam2p, const float* __restrict__ means,
                               float* __restrict__ Jscat, float* __restrict__ uo,
                               float* __restrict__ vo, float* __restrict__ w1o,
                               float* __restrict__ w2o) {
  long t = (long)blockIdx.x * blockDim.x + threadIdx.x;
  if (t >= (long)Bn * HW) return;
  int  b  = (int)(t / HW);
  long px = t % HW;
  float a0 = means[b * 3 + 0], a1 = means[b * 3 + 1], a2 = means[b * 3 + 2];
  int i0 = 0, i1 = 1, i2 = 2;
  // stable ascending sort of 3 (strict < keeps tie order == jnp.argsort)
  if (a1 < a0) { float tm = a0; a0 = a1; a1 = tm; int ti = i0; i0 = i1; i1 = ti; }
  if (a2 < a1) { float tm = a1; a1 = a2; a2 = tm; int ti = i1; i1 = i2; i2 = ti; }
  if (a1 < a0) { float tm = a0; a0 = a1; a1 = tm; int ti = i0; i0 = i1; i1 = ti; }
  const float lam1 = lam1p[0], lam2 = lam2p[0];
  const long base = (long)b * 3 * HW + px;
  const long b1   = (long)b * HW + px;
  const float Jl  = J[base + (long)i2 * HW];
  const float Jm  = J[base + (long)i1 * HW];
  const float Js  = J[base + (long)i0 * HW];
  const float uu = u[b1], vv = v[b1], ww1 = w1[b1], ww2 = w2[b1];
  const float Jlbar = a2;
  const float Jmbar = Jl - uu + ww1 / lam1;
  const float Jsbar = Jl - vv + ww2 / lam2;
  Jscat[base + (long)i2 * HW] = Jl;
  Jscat[base + (long)i1 * HW] = Jm + (Jlbar - Jmbar) * Jl;
  Jscat[base + (long)i0 * HW] = Js + (Jlbar - Jsbar) * Jl;
  const float x1 = Jlbar - Jmbar + ww1 / lam1;
  const float un = softthr(x1, 1.f / lam1);
  const float x2 = Jlbar - Jsbar + ww2 / lam2;
  const float vn = softthr(x2, 1.f / lam2);
  uo[b1] = un; vo[b1] = vn;
  w1o[b1] = ww1 + lam1 * (Jlbar - Jmbar - un);
  w2o[b1] = ww2 + lam2 * (Jlbar - Jsbar - vn);
}

// ------------------------------------------------------------------
// per-(b,c) spatial mean of B-update, value computed on the fly
__global__ void bmean_kernel(const float* __restrict__ Jscat, const float* __restrict__ t,
                             const float* __restrict__ I, const float* __restrict__ Bp,
                             float* __restrict__ bm) {
  __shared__ float sm[256];
  const long base = (long)blockIdx.x * HW;
  float s = 0.f;
  for (int i = threadIdx.x; i < HW; i += 256) {
    const float tv = t[base + i];
    const float om = 1.f - tv;
    const float num = GAMMA_1 * Bp[base + i] - (Jscat[base + i] * tv - I[base + i]) * om;
    s += num / (om * om + GAMMA_1);
  }
  sm[threadIdx.x] = s; __syncthreads();
  for (int o = 128; o > 0; o >>= 1) {
    if (threadIdx.x < o) sm[threadIdx.x] += sm[threadIdx.x + o];
    __syncthreads();
  }
  if (threadIdx.x == 0) bm[blockIdx.x] = sm[0] / (float)HW;
}

// ------------------------------------------------------------------
// t update + 1x1 conv/ReLU + J,Q,B,Y outputs + f16 padded RDN input
__global__ void tjq_kernel(const float* __restrict__ I, const float* __restrict__ tp,
                           const float* __restrict__ Yv, const float* __restrict__ Zin,
                           const float* __restrict__ Qin, const float* __restrict__ Rin,
                           const float* __restrict__ Jscat, const float* __restrict__ bm,
                           const float* __restrict__ betap, const float* __restrict__ etap,
                           const float* __restrict__ t1dw, float* __restrict__ out,
                           half_t* __restrict__ X32) {
  long t = (long)blockIdx.x * blockDim.x + threadIdx.x;
  if (t >= (long)Bn * HW) return;
  int  b  = (int)(t / HW);
  long px = t % HW;
  const float beta = betap[0], eta = etap[0];
  float tpre[3];
  #pragma unroll
  for (int c = 0; c < 3; ++c) {
    const long o = ((long)b * 3 + c) * HW + px;
    const float Bm = bm[b * 3 + c];
    const float Jv = Jscat[o];
    const float num = GAMMA_2 * tp[o] + eta * Zin[o] - Rin[o] - (Bm - I[o]) * (Jv - Bm);
    const float den = (Jv - Bm) * (Jv - Bm) + GAMMA_2 + eta;
    tpre[c] = num / den;
  }
  float t1 = t1dw[0] * tpre[0] + t1dw[1] * tpre[1] + t1dw[2] * tpre[2];
  t1 = t1 > 0.f ? t1 : 0.f;
  union { half_t h[32]; v8h v[4]; } xu;
  #pragma unroll
  for (int c = 0; c < 32; ++c) xu.h[c] = (half_t)0.f;
  #pragma unroll
  for (int c = 0; c < 3; ++c) {
    const long o = ((long)b * 3 + c) * HW + px;
    const float Bm = bm[b * 3 + c];
    const float Yy = Yv[o], Qq = Qin[o], Ii = I[o];
    const float J2 = (beta * Yy - Qq - (Bm * (1.f - t1) - Ii) * t1) / (t1 * t1 + beta);
    out[o]          = Bm;                       // B
    out[N3 + o]     = t1;                       // t
    out[2 * N3 + o] = J2;                       // J
    out[3 * N3 + o] = Yy;                       // Y (passthrough)
    out[5 * N3 + o] = Qq + beta * (J2 - Yy);    // Q
    xu.h[c] = (half_t)(t1 + Rin[o] / eta);      // RDN input
  }
  v8h* xd = (v8h*)(X32 + t * 32);
  #pragma unroll
  for (int k = 0; k < 4; ++k) xd[k] = xu.v[k];
  if (t == 0) out[7 * N3 + 4 * N1] = beta;      // beta (passthrough)
}

// ------------------------------------------------------------------
// pack OIHW fp32 weights -> f16 [CoutPad][KT+8] (K = tap*CINP+cin),
// row stride padded by 8 halfs to kill LDS bank conflicts on B reads.
__global__ void pack_w(const float* __restrict__ w, half_t* __restrict__ dst,
                       int Cout, int Cin, int CoutPad, int CINP, int TAPS) {
  const int KT  = TAPS * CINP;
  const int KTP = KT + 8;
  const int i = blockIdx.x * blockDim.x + threadIdx.x;
  const int tot = CoutPad * KTP;
  if (i >= tot) return;
  const int o   = i / KTP;
  const int col = i % KTP;
  float v = 0.f;
  if (col < KT) {
    const int tap = col / CINP;
    const int ci  = col % CINP;
    if (o < Cout && ci < Cin) v = w[(o * Cin + ci) * TAPS + tap];
  }
  dst[i] = (half_t)v;
}

// ------------------------------------------------------------------
// copy 64 channels per pixel between strided NHWC f16 buffers
__global__ void copy_ch(const half_t* __restrict__ src, int sStride, int sOff,
                        half_t* __restrict__ dst, int dStride, int dOff) {
  const long t = (long)blockIdx.x * blockDim.x + threadIdx.x;   // Bn*HW*8 threads
  const long pix = t >> 3;
  const int  c8  = (int)(t & 7) * 8;
  *(v8h*)(dst + pix * (long)dStride + dOff + c8) =
      *(const v8h*)(src + pix * (long)sStride + sOff + c8);
}

// ------------------------------------------------------------------
// implicit-GEMM conv via v_wmma_f32_16x16x32_f16.
// Block = 8 waves, 128 pixels of one row; each wave: 16 pixels x NT*16 couts.
// Weights staged global->LDS with gfx1250 async-to-LDS ops, then fed to WMMA
// B-operands via ds_load; A-operands streamed from global (NHWC, contiguous K).
// A layout (16-bit 16x32): lane<16 -> K 0-7 & 16-23 ; lane>=16 -> K 8-15 & 24-31.
// B layout (16-bit 32x16): lane holds N=lane&15, same K order as A.
template<int NT, int TAPS, int CINP, bool RELU, bool ADD, bool FINAL>
__global__ __launch_bounds__(256) void conv_wmma(
    const half_t* __restrict__ in, int inStride,
    const half_t* __restrict__ wp, const float* __restrict__ bias,
    const half_t* __restrict__ addsrc, int addStride, int addOff,
    half_t* __restrict__ outp, int outStride, int outOff,
    float* __restrict__ zout, float* __restrict__ rout,
    const float* __restrict__ tin, const float* __restrict__ rin,
    const float* __restrict__ etap) {
  constexpr int KT   = TAPS * CINP;
  constexpr int KTP  = KT + 8;        // padded LDS row stride (halfs)
  constexpr int ROWS = NT * 16;
  extern __shared__ half_t wlds[];    // ROWS * KTP halfs

  // ---- stage packed weights into LDS (async global->LDS, 16B chunks) ----
  {
    const int chunks = ROWS * KTP / 8;
    for (int i = threadIdx.x; i < chunks; i += 256) {
      const unsigned ldsoff = (unsigned)(i * 16);                    // byte offset
      const unsigned long long ga =
          (unsigned long long)(uintptr_t)(wp + (long)i * 8);
      asm volatile("global_load_async_to_lds_b128 %0, %1, off"
                   :: "v"(ldsoff), "v"(ga) : "memory");
    }
    asm volatile("s_wait_asynccnt 0" ::: "memory");
    __syncthreads();
  }

  const int lane  = threadIdx.x & 31;
  const int wave  = threadIdx.x >> 5;
  const int b     = blockIdx.z;
  const int y     = blockIdx.y;
  const int xbase = blockIdx.x * 128 + wave * 16;
  const int row   = lane & 15;
  const int khi   = lane >> 4;
  const int xA    = xbase + row;

  v8f acc[NT] = {};

  #pragma unroll
  for (int tap = 0; tap < TAPS; ++tap) {
    const int dy = (TAPS == 9) ? (tap / 3 - 1) : 0;
    const int dx = (TAPS == 9) ? (tap % 3 - 1) : 0;
    const int yy = y + dy;
    const int xx = xA + dx;
    const bool inb = (yy >= 0) & (yy < Hh) & (xx >= 0) & (xx < Ww);
    const long pbase = ((long)(b * Hh + yy) * Ww + xx) * (long)inStride;
    #pragma unroll
    for (int c0 = 0; c0 < CINP; c0 += 32) {
      v16h a = {};
      if (inb) {
        const half_t* ap = in + pbase + c0 + khi * 8;
        union { v16h v; v8h h[2]; } au;
        au.h[0] = *(const v8h*)ap;
        au.h[1] = *(const v8h*)(ap + 16);
        a = au.v;
      }
      const int kb = tap * CINP + c0 + khi * 8;
      #pragma unroll
      for (int nt = 0; nt < NT; ++nt) {
        const int n = nt * 16 + row;
        const half_t* bp = wlds + n * KTP + kb;
        union { v16h v; v8h h[2]; } bu;
        bu.h[0] = *(const v8h*)bp;
        bu.h[1] = *(const v8h*)(bp + 16);
        acc[nt] = __builtin_amdgcn_wmma_f32_16x16x32_f16(false, a, false, bu.v,
                                                         (short)0, acc[nt], false, false);
      }
    }
  }

  #pragma unroll
  for (int nt = 0; nt < NT; ++nt) {
    const int cout = nt * 16 + row;
    #pragma unroll
    for (int r = 0; r < 8; ++r) {
      const int px  = xbase + khi * 8 + r;
      const long pix = (long)(b * Hh + y) * Ww + px;
      float vv = acc[nt][r];
      if constexpr (!FINAL) {
        vv += bias[cout];
        if constexpr (ADD) vv += (float)addsrc[pix * (long)addStride + addOff + cout];
        if constexpr (RELU) vv = vv > 0.f ? vv : 0.f;
        outp[pix * (long)outStride + outOff + cout] = (half_t)vv;
      } else {
        if (cout < 3) {
          vv += bias[cout];
          const long o = ((long)(b * 3 + cout) * Hh + y) * Ww + px;
          zout[o] = vv;                                   // Z
          rout[o] = rin[o] + etap[0] * (tin[o] - vv);     // R = R + eta*(t - Z)
        }
      }
    }
  }
}

// ------------------------------------------------------------------
extern "C" void kernel_launch(void* const* d_in, const int* in_sizes, int n_in,
                              void* d_out, int out_size, void* d_ws, size_t ws_size,
                              hipStream_t stream) {
  (void)in_sizes; (void)n_in; (void)out_size; (void)ws_size;
  const float* I    = (const float*)d_in[0];
  const float* t_p  = (const float*)d_in[1];
  const float* B_p  = (const float*)d_in[2];
  const float* t_in = (const float*)d_in[4];
  const float* J_in = (const float*)d_in[5];
  const float* Y    = (const float*)d_in[6];
  const float* Z_in = (const float*)d_in[7];
  const float* Q    = (const float*)d_in[8];
  const float* R    = (const float*)d_in[9];
  const float* u    = (const float*)d_in[10];
  const float* v    = (const float*)d_in[11];
  const float* w1   = (const float*)d_in[12];
  const float* w2   = (const float*)d_in[13];
  const float* beta = (const float*)d_in[14];
  const float* eta  = (const float*)d_in[15];
  const float* lam1 = (const float*)d_in[16];
  const float* lam2 = (const float*)d_in[17];
  const float* t1d  = (const float*)d_in[18];
  const float* sfe1w = (const float*)d_in[19]; const float* sfe1b = (const float*)d_in[20];
  const float* sfe2w = (const float*)d_in[21]; const float* sfe2b = (const float*)d_in[22];
  const float* r0dw  = (const float*)d_in[23]; const float* r0db  = (const float*)d_in[24];
  const float* r0lw  = (const float*)d_in[25]; const float* r0lb  = (const float*)d_in[26];
  const float* r1dw  = (const float*)d_in[27]; const float* r1db  = (const float*)d_in[28];
  const float* r1lw  = (const float*)d_in[29]; const float* r1lb  = (const float*)d_in[30];
  const float* gf1w  = (const float*)d_in[31]; const float* gf1b  = (const float*)d_in[32];
  const float* gf2w  = (const float*)d_in[33]; const float* gf2b  = (const float*)d_in[34];
  const float* outw  = (const float*)d_in[35]; const float* outb  = (const float*)d_in[36];

  char*  ws  = (char*)d_ws;
  float* out = (float*)d_out;
  float* meansJ = (float*)(ws + 0);
  float* bmeans = (float*)(ws + 128);
  // packed weights, padded row stride KT+8 halfs (all offsets 64B aligned)
  size_t po = 256;
  half_t* sfe1p = (half_t*)(ws + po); po += 37888;   // 64 x 296
  half_t* sfe2p = (half_t*)(ws + po); po += 74752;   // 64 x 584
  half_t* r0dp  = (half_t*)(ws + po); po += 74752;
  half_t* r0lp  = (half_t*)(ws + po); po += 17408;   // 64 x 136
  half_t* r1dp  = (half_t*)(ws + po); po += 74752;
  half_t* r1lp  = (half_t*)(ws + po); po += 17408;
  half_t* gf1p  = (half_t*)(ws + po); po += 17408;
  half_t* gf2p  = (half_t*)(ws + po); po += 74752;
  half_t* outp  = (half_t*)(ws + po); po += 18688;   // 16 x 584

  const size_t offA = (size_t)1 << 20;
  const size_t szA  = (size_t)Bn * HW * 32 * 2;      // 33.5 MB
  const size_t szB  = (size_t)Bn * HW * 64 * 2;      // 67  MB
  half_t* X32  = (half_t*)(ws + offA);
  half_t* SFE1 = (half_t*)(ws + offA + szA);
  half_t* Cbuf = (half_t*)(ws + offA + szA + szB);
  half_t* Dbuf = (half_t*)(ws + offA + szA + 2 * szB);   // 128-ch region (2*szB)
  half_t* Ebuf = (half_t*)(ws + offA + szA + 4 * szB);   // 128-ch region (2*szB)
  float*  Jscat = (float*)Dbuf;  // fp32 alias, dead before Dbuf is first written

  const dim3 b256(256);
  const dim3 eg(2048);                    // Bn*HW / 256
  const dim3 cg(Ww / 128, Hh, Bn);        // conv grid
  const size_t sh964 = 64 * 584 * 2;      // 74752 B  (<4,9,64>)
  const size_t sh932 = 64 * 296 * 2;      // 37888 B  (<4,9,32>)
  const size_t sh1   = 64 * 136 * 2;      // 17408 B  (<4,1,128>)
  const size_t shF   = 16 * 584 * 2;      // 18688 B  (<1,9,64>)

  means_kernel<<<24, b256, 0, stream>>>(J_in, meansJ);
  scatter_kernel<<<eg, b256, 0, stream>>>(J_in, u, v, w1, w2, lam1, lam2, meansJ, Jscat,
                                          out + 7 * N3, out + 7 * N3 + N1,
                                          out + 7 * N3 + 2 * N1, out + 7 * N3 + 3 * N1);
  bmean_kernel<<<24, b256, 0, stream>>>(Jscat, t_in, I, B_p, bmeans);
  tjq_kernel<<<eg, b256, 0, stream>>>(I, t_p, Y, Z_in, Q, R, Jscat, bmeans,
                                      beta, eta, t1d, out, X32);

  pack_w<<<74,  b256, 0, stream>>>(sfe1w, sfe1p, 64, 3,   64, 32,  9);
  pack_w<<<146, b256, 0, stream>>>(sfe2w, sfe2p, 64, 64,  64, 64,  9);
  pack_w<<<146, b256, 0, stream>>>(r0dw,  r0dp,  64, 64,  64, 64,  9);
  pack_w<<<34,  b256, 0, stream>>>(r0lw,  r0lp,  64, 128, 64, 128, 1);
  pack_w<<<146, b256, 0, stream>>>(r1dw,  r1dp,  64, 64,  64, 64,  9);
  pack_w<<<34,  b256, 0, stream>>>(r1lw,  r1lp,  64, 128, 64, 128, 1);
  pack_w<<<34,  b256, 0, stream>>>(gf1w,  gf1p,  64, 128, 64, 128, 1);
  pack_w<<<146, b256, 0, stream>>>(gf2w,  gf2p,  64, 64,  64, 64,  9);
  pack_w<<<37,  b256, 0, stream>>>(outw,  outp,  3,  64,  16, 64,  9);

  // sfe1: X32(32) -> SFE1(64)
  conv_wmma<4, 9, 32, false, false, false><<<cg, b256, sh932, stream>>>(
      X32, 32, sfe1p, sfe1b, nullptr, 0, 0, SFE1, 64, 0,
      nullptr, nullptr, nullptr, nullptr, nullptr);
  // sfe2: SFE1 -> Cbuf (h0)
  conv_wmma<4, 9, 64, false, false, false><<<cg, b256, sh964, stream>>>(
      SFE1, 64, sfe2p, sfe2b, nullptr, 0, 0, Cbuf, 64, 0,
      nullptr, nullptr, nullptr, nullptr, nullptr);
  // RDB0: D = [h0, relu(conv3(h0))]
  copy_ch<<<16384, b256, 0, stream>>>(Cbuf, 64, 0, Dbuf, 128, 0);
  conv_wmma<4, 9, 64, true, false, false><<<cg, b256, sh964, stream>>>(
      Cbuf, 64, r0dp, r0db, nullptr, 0, 0, Dbuf, 128, 64,
      nullptr, nullptr, nullptr, nullptr, nullptr);
  // h1 = h0 + lff(D) -> Ebuf[:,:64]
  conv_wmma<4, 1, 128, false, true, false><<<cg, b256, sh1, stream>>>(
      Dbuf, 128, r0lp, r0lb, Cbuf, 64, 0, Ebuf, 128, 0,
      nullptr, nullptr, nullptr, nullptr, nullptr);
  // RDB1: D = [h1, relu(conv3(h1))]
  copy_ch<<<16384, b256, 0, stream>>>(Ebuf, 128, 0, Dbuf, 128, 0);
  conv_wmma<4, 9, 64, true, false, false><<<cg, b256, sh964, stream>>>(
      Ebuf, 128, r1dp, r1db, nullptr, 0, 0, Dbuf, 128, 64,
      nullptr, nullptr, nullptr, nullptr, nullptr);
  // h2 = h1 + lff(D) -> Ebuf[:,64:]
  conv_wmma<4, 1, 128, false, true, false><<<cg, b256, sh1, stream>>>(
      Dbuf, 128, r1lp, r1lb, Ebuf, 128, 0, Ebuf, 128, 64,
      nullptr, nullptr, nullptr, nullptr, nullptr);
  // g = gff1(concat feats) -> Cbuf
  conv_wmma<4, 1, 128, false, false, false><<<cg, b256, sh1, stream>>>(
      Ebuf, 128, gf1p, gf1b, nullptr, 0, 0, Cbuf, 64, 0,
      nullptr, nullptr, nullptr, nullptr, nullptr);
  // g2 = gff2(g) + sfe1 -> Dbuf (as 64-stride)
  conv_wmma<4, 9, 64, false, true, false><<<cg, b256, sh964, stream>>>(
      Cbuf, 64, gf2p, gf2b, SFE1, 64, 0, Dbuf, 64, 0,
      nullptr, nullptr, nullptr, nullptr, nullptr);
  // Z = out(g2); fused R = R + eta*(t - Z); fp32 NCHW stores
  conv_wmma<1, 9, 64, false, false, true><<<cg, b256, shF, stream>>>(
      Dbuf, 64, outp, outb, nullptr, 0, 0, nullptr, 0, 0,
      out + 4 * N3, out + 6 * N3, out + N3, R, eta);
}
